// MarketBDH_79448305041637
// MI455X (gfx1250) — compile-verified, matrix operand order checked by reading, not run on hardware
//
#include <hip/hip_runtime.h>

// ---------------------------------------------------------------------------
// BDH forward for MI455X (gfx1250):
//   bf16 WMMA GEMM cores (v_wmma_f32_16x16x32_bf16), TDM (tensor_load_to_lds)
//   double-buffered tile staging, f32 VALU epilogues (LN / RoPE / gating).
// ---------------------------------------------------------------------------

typedef __attribute__((ext_vector_type(16))) __bf16 v16bf;
typedef __attribute__((ext_vector_type(8)))  __bf16 v8bf;
typedef __attribute__((ext_vector_type(8)))  float  v8f;
typedef __attribute__((ext_vector_type(4)))  unsigned int u32x4;
typedef __attribute__((ext_vector_type(8)))  int i32x8;
typedef __attribute__((ext_vector_type(4)))  int i32x4;

#define kB   2
#define kT   1024
#define kNH  8
#define kD   512
#define kNI  2048      // sparse code dim per head
#define kNL  3
#define kIN  372       // NA*F
#define kOUT 186       // NA*NUM_CLASSES

#define BM 128
#define BN 128
#define BK 32
#define A_TILE_BYTES (BM * BK * 2)                    // 8 KB
#define B_TILE_BYTES (BN * BK * 2)                    // 8 KB
#define SMEM_BYTES   (2 * A_TILE_BYTES + 2 * B_TILE_BYTES)  // 32 KB

// ---------------------------------------------------------------------------
// Tensor Data Mover: async 2D tile load global -> LDS (ISA 08_async_tensor §8)
// ---------------------------------------------------------------------------
static __device__ __forceinline__ void tdm_ll(u32x4 g0, i32x8 g1)
{
    i32x4 gz = {0, 0, 0, 0};
#if defined(__clang_major__) && (__clang_major__ >= 23)
    i32x8 g4 = {0, 0, 0, 0, 0, 0, 0, 0};
    __builtin_amdgcn_tensor_load_to_lds(g0, g1, gz, gz, g4, 0);
#else
    __builtin_amdgcn_tensor_load_to_lds(g0, g1, gz, gz, 0);
#endif
}

// Loads a [tileRows x tileK] bf16 tile (row stride = strideElems elements)
// starting at gptr into LDS byte offset ldsOff, row-major contiguous.
static __device__ __forceinline__ void tdm_tile_load(
    const __bf16* gptr, unsigned ldsOff, int tileK, int tileRows,
    int strideElems, int tensorK, int tensorRows)
{
    const unsigned long long ga = (unsigned long long)(uintptr_t)gptr;
    u32x4 g0;
    g0.x = 1u;                                        // count=1, user descriptor
    g0.y = ldsOff;                                    // lds_addr (bytes)
    g0.z = (unsigned)(ga & 0xFFFFFFFFull);            // global_addr[31:0]
    g0.w = (unsigned)((ga >> 32) & 0x01FFFFFFull)     // global_addr[56:32]
         | (2u << 30);                                // type = 2 ("image")
    i32x8 g1;
    g1[0] = 0x10000;                                  // data_size=1 (2B), mask=0
    g1[1] = (int)(((unsigned)tensorK & 0xFFFFu) << 16);          // tensor_dim0[15:0]
    g1[2] = (int)((((unsigned)tensorK >> 16) & 0xFFFFu)          // tensor_dim0[31:16]
         | (((unsigned)tensorRows & 0xFFFFu) << 16));            // tensor_dim1[15:0]
    g1[3] = (int)((((unsigned)tensorRows >> 16) & 0xFFFFu)       // tensor_dim1[31:16]
         | (((unsigned)tileK & 0xFFFFu) << 16));                 // tile_dim0
    g1[4] = (int)((unsigned)tileRows & 0xFFFFu);                 // tile_dim1 (tile_dim2=0)
    g1[5] = strideElems;                                         // dim0_stride[31:0]
    g1[6] = 0;                                                   // dim0_stride[47:32]
    g1[7] = 0;
    tdm_ll(g0, g1);
}

static __device__ __forceinline__ void tdm_wait()
{
    __builtin_amdgcn_s_wait_tensorcnt((short)0);
}

// ---------------------------------------------------------------------------
// Generic bf16 GEMM:  C[M,N] = A[M,K] @ B[K,N]   (row-major, f32 accumulate)
// Block tile 128x128x32, 8 wave32 waves each computing 32x64 (8 WMMA/step).
// A staged by TDM (double buffered). B: TDM when TRANSB (contiguous tile),
// else manual register-double-buffered transpose scatter.
// EPI: 0 = f32 store, 1 = relu->bf16, 2 = relu*gate->bf16,
//      3 = strict causal mask (col < row) -> bf16
// Requires M%128==0, N%128==0, K%32==0 (all call sites satisfy this).
// ---------------------------------------------------------------------------
template<int EPI, bool TRANSB>
__global__ __launch_bounds__(256)
void gemm128_wmma(const __bf16* __restrict__ A, const __bf16* __restrict__ Bm,
                  float* __restrict__ C32, __bf16* __restrict__ Cbf,
                  const __bf16* __restrict__ gate,
                  int M, int N, int K, int lda, int ldb, int ldc, int ldg)
{
    extern __shared__ __align__(16) char smem[];
    __bf16 (*As)[BM][BK] = reinterpret_cast<__bf16 (*)[BM][BK]>(smem);
    __bf16 (*Bs)[BN][BK] = reinterpret_cast<__bf16 (*)[BN][BK]>(smem + 2 * A_TILE_BYTES);

    const int tid  = threadIdx.x;
    const int lane = tid & 31;
    const int wid  = tid >> 5;          // 8 waves
    const int hi   = lane >> 4;
    const int l16  = lane & 15;

    const int rowBase = blockIdx.y * BM;
    const int colBase = blockIdx.x * BN;
    const int wr = wid & 3;             // wave row group: 32 rows
    const int wc = wid >> 2;            // wave col group: 64 cols

    const int nk = K / BK;
    v8f acc[2][4] = {};

    // ---- prologue: stage tile 0 ------------------------------------------
    if (wid == 0)
        tdm_tile_load(A + (long)rowBase * lda, 0u, BK, BM, lda, K, M);
    if (TRANSB) {
        if (wid == 1)
            tdm_tile_load(Bm + (long)colBase * ldb, (unsigned)(2 * A_TILE_BYTES),
                          BK, BN, ldb, K, N);
    } else {
        const int c0 = tid, c1 = tid + 256;
        union { uint4 u; __bf16 h[8]; } v0, v1;
        v0.u = *(const uint4*)(Bm + (long)(c0 >> 4) * ldb + colBase + (c0 & 15) * 8);
        v1.u = *(const uint4*)(Bm + (long)(c1 >> 4) * ldb + colBase + (c1 & 15) * 8);
#pragma unroll
        for (int i = 0; i < 8; ++i) Bs[0][(c0 & 15) * 8 + i][c0 >> 4] = v0.h[i];
#pragma unroll
        for (int i = 0; i < 8; ++i) Bs[0][(c1 & 15) * 8 + i][c1 >> 4] = v1.h[i];
    }
    if (wid < (TRANSB ? 2 : 1)) tdm_wait();
    __syncthreads();

    // ---- main loop --------------------------------------------------------
    for (int kt = 0; kt < nk; ++kt) {
        const int cur = kt & 1, nxt = cur ^ 1;
        const bool more = (kt + 1) < nk;
        uint4 r0 = {0, 0, 0, 0}, r1 = {0, 0, 0, 0};

        if (more) {
            const int k1 = (kt + 1) * BK;
            if (wid == 0)
                tdm_tile_load(A + (long)rowBase * lda + k1,
                              (unsigned)(nxt * A_TILE_BYTES), BK, BM, lda, K, M);
            if (TRANSB) {
                if (wid == 1)
                    tdm_tile_load(Bm + (long)colBase * ldb + k1,
                                  (unsigned)(2 * A_TILE_BYTES + nxt * B_TILE_BYTES),
                                  BK, BN, ldb, K, N);
            } else {
                const int c0 = tid, c1 = tid + 256;
                r0 = *(const uint4*)(Bm + (long)(k1 + (c0 >> 4)) * ldb + colBase + (c0 & 15) * 8);
                r1 = *(const uint4*)(Bm + (long)(k1 + (c1 >> 4)) * ldb + colBase + (c1 & 15) * 8);
            }
        }

        // ---- compute from buffer `cur` (ISA 7.12.2 fragment layouts) -----
        v16bf afr0, afr1;
        {
            const int ar0 = (wr * 2) * 16 + l16;
            v8bf x0 = *(const v8bf*)(&As[cur][ar0][hi * 8]);
            v8bf x1 = *(const v8bf*)(&As[cur][ar0][16 + hi * 8]);
            afr0 = __builtin_shufflevector(x0, x1,
                0,1,2,3,4,5,6,7,8,9,10,11,12,13,14,15);
            const int ar1 = ar0 + 16;
            v8bf y0 = *(const v8bf*)(&As[cur][ar1][hi * 8]);
            v8bf y1 = *(const v8bf*)(&As[cur][ar1][16 + hi * 8]);
            afr1 = __builtin_shufflevector(y0, y1,
                0,1,2,3,4,5,6,7,8,9,10,11,12,13,14,15);
        }
#pragma unroll
        for (int j = 0; j < 4; ++j) {
            const int bc = (wc * 4 + j) * 16 + l16;
            v8bf b0 = *(const v8bf*)(&Bs[cur][bc][hi * 8]);
            v8bf b1 = *(const v8bf*)(&Bs[cur][bc][16 + hi * 8]);
            v16bf bfr = __builtin_shufflevector(b0, b1,
                0,1,2,3,4,5,6,7,8,9,10,11,12,13,14,15);
            acc[0][j] = __builtin_amdgcn_wmma_f32_16x16x32_bf16(
                false, afr0, false, bfr, (short)0, acc[0][j], false, false);
            acc[1][j] = __builtin_amdgcn_wmma_f32_16x16x32_bf16(
                false, afr1, false, bfr, (short)0, acc[1][j], false, false);
        }

        if (more && !TRANSB) {
            const int c0 = tid, c1 = tid + 256;
            union { uint4 u; __bf16 h[8]; } v0, v1;
            v0.u = r0; v1.u = r1;
#pragma unroll
            for (int i = 0; i < 8; ++i) Bs[nxt][(c0 & 15) * 8 + i][c0 >> 4] = v0.h[i];
#pragma unroll
            for (int i = 0; i < 8; ++i) Bs[nxt][(c1 & 15) * 8 + i][c1 >> 4] = v1.h[i];
        }
        if (more && (wid < (TRANSB ? 2 : 1))) tdm_wait();
        __syncthreads();
    }

    // ---- epilogue: C elem v @ lane L -> (M = v + 8*hi, N = l16) ----------
#pragma unroll
    for (int i = 0; i < 2; ++i)
#pragma unroll
    for (int j = 0; j < 4; ++j)
#pragma unroll
    for (int v = 0; v < 8; ++v) {
        const int gr = rowBase + (wr * 2 + i) * 16 + v + hi * 8;
        const int gc = colBase + (wc * 4 + j) * 16 + l16;
        const float xv = acc[i][j][v];
        if (EPI == 0) {
            C32[(long)gr * ldc + gc] = xv;
        } else if (EPI == 1) {
            Cbf[(long)gr * ldc + gc] = (__bf16)(xv > 0.f ? xv : 0.f);
        } else if (EPI == 2) {
            const float g = (float)gate[(long)gr * ldg + gc];
            Cbf[(long)gr * ldc + gc] = (__bf16)((xv > 0.f ? xv : 0.f) * g);
        } else {
            Cbf[(long)gr * ldc + gc] = (__bf16)(gc < gr ? xv : 0.f);
        }
    }
}

// ---------------------------------------------------------------------------
// Helpers
// ---------------------------------------------------------------------------
__device__ __forceinline__ float block_sum(float v, float* red)
{
    const int tid = threadIdx.x;
    red[tid] = v;
    __syncthreads();
#pragma unroll
    for (int s = 128; s > 0; s >>= 1) {
        if (tid < s) red[tid] += red[tid + s];
        __syncthreads();
    }
    float r = red[0];
    __syncthreads();
    return r;
}

__global__ __launch_bounds__(256)
void f2bf_kernel(const float* __restrict__ in, __bf16* __restrict__ out, long n)
{
    long i = (long)blockIdx.x * 256 + threadIdx.x;
    if (i < n) out[i] = (__bf16)in[i];
}

// x[B*T,372] @ Wi[372,512] + bi  ->  out f32 [B*T,512]
__global__ __launch_bounds__(256)
void inproj_kernel(const float* __restrict__ x, const float* __restrict__ Wi,
                   const float* __restrict__ bi, float* __restrict__ out)
{
    __shared__ float xs[kIN];
    const long r = blockIdx.x;
    for (int i = threadIdx.x; i < kIN; i += 256) xs[i] = x[r * kIN + i];
    __syncthreads();
    const int c0 = threadIdx.x, c1 = threadIdx.x + 256;
    float a0 = 0.f, a1 = 0.f;
    for (int k = 0; k < kIN; ++k) {
        const float xv = xs[k];
        a0 += xv * Wi[(long)k * kD + c0];
        a1 += xv * Wi[(long)k * kD + c1];
    }
    out[r * kD + c0] = a0 + bi[c0];
    out[r * kD + c1] = a1 + bi[c1];
}

// Row-wise LayerNorm over 512, biased var, eps=1e-5; optional f32/bf16 outs.
__global__ __launch_bounds__(256)
void ln512_kernel(const float* __restrict__ in, float* __restrict__ out32,
                  __bf16* __restrict__ outbf)
{
    __shared__ float red[256];
    const long r = blockIdx.x;
    const int tid = threadIdx.x;
    const float v0 = in[r * kD + tid];
    const float v1 = in[r * kD + tid + 256];
    const float mean = block_sum(v0 + v1, red) * (1.0f / kD);
    const float d0 = v0 - mean, d1 = v1 - mean;
    const float var = block_sum(d0 * d0 + d1 * d1, red) * (1.0f / kD);
    const float inv = rsqrtf(var + 1e-5f);
    const float o0 = d0 * inv, o1 = d1 * inv;
    if (out32) { out32[r * kD + tid] = o0; out32[r * kD + tid + 256] = o1; }
    if (outbf) {
        outbf[r * kD + tid]       = (__bf16)o0;
        outbf[r * kD + tid + 256] = (__bf16)o1;
    }
}

// h = LN(h + LN(t));  writes h32 (f32) and hb (bf16)
__global__ __launch_bounds__(256)
void lnresln_kernel(const float* __restrict__ t32, float* __restrict__ h32,
                    __bf16* __restrict__ hb)
{
    __shared__ float red[256];
    const long r = blockIdx.x;
    const int tid = threadIdx.x;
    float v0 = t32[r * kD + tid];
    float v1 = t32[r * kD + tid + 256];
    float mean = block_sum(v0 + v1, red) * (1.0f / kD);
    float d0 = v0 - mean, d1 = v1 - mean;
    float var = block_sum(d0 * d0 + d1 * d1, red) * (1.0f / kD);
    float inv = rsqrtf(var + 1e-5f);
    float w0 = h32[r * kD + tid]       + d0 * inv;
    float w1 = h32[r * kD + tid + 256] + d1 * inv;
    mean = block_sum(w0 + w1, red) * (1.0f / kD);
    d0 = w0 - mean; d1 = w1 - mean;
    var = block_sum(d0 * d0 + d1 * d1, red) * (1.0f / kD);
    inv = rsqrtf(var + 1e-5f);
    const float z0 = d0 * inv, z1 = d1 * inv;
    h32[r * kD + tid]       = z0;
    h32[r * kD + tid + 256] = z1;
    hb[r * kD + tid]        = (__bf16)z0;
    hb[r * kD + tid + 256]  = (__bf16)z1;
}

// RoPE (theta = 2^16): pair (n, n+1024), freq = 2^(-n/64), angle = t*freq.
__global__ __launch_bounds__(256)
void rope_kernel(const __bf16* __restrict__ hsp, __bf16* __restrict__ qr)
{
    const long row = blockIdx.x;                     // B*NH*T rows
    const int  n   = blockIdx.y * 256 + threadIdx.x; // 0..1023
    const int  t   = (int)(row & (kT - 1));
    const float f  = exp2f(-(float)n * (1.0f / 64.0f));
    const float ph = (float)t * f;
    const float c  = cosf(ph);
    const float s  = sinf(ph);
    const long base = row * kNI;
    const float q1 = (float)hsp[base + n];
    const float q2 = (float)hsp[base + n + kNI / 2];
    qr[base + n]           = (__bf16)(q1 * c - q2 * s);
    qr[base + n + kNI / 2] = (__bf16)(q2 * c + q1 * s);
}

// out[B*T,186] = h[B*T,512] @ Wo[512,186] + bo
__global__ __launch_bounds__(256)
void outproj_kernel(const float* __restrict__ h, const float* __restrict__ Wo,
                    const float* __restrict__ bo, float* __restrict__ out)
{
    __shared__ float hs[kD];
    const long r = blockIdx.x;
    const int tid = threadIdx.x;
    hs[tid]       = h[r * kD + tid];
    hs[tid + 256] = h[r * kD + tid + 256];
    __syncthreads();
    if (tid < kOUT) {
        float a = bo[tid];
        for (int k = 0; k < kD; ++k) a += hs[k] * Wo[(long)k * kOUT + tid];
        out[r * kOUT + tid] = a;
    }
}

// ---------------------------------------------------------------------------
// Host orchestration
// ---------------------------------------------------------------------------
extern "C" void kernel_launch(void* const* d_in, const int* in_sizes, int n_in,
                              void* d_out, int out_size, void* d_ws, size_t ws_size,
                              hipStream_t stream)
{
    (void)in_sizes; (void)n_in; (void)out_size; (void)ws_size;

    const float* x     = (const float*)d_in[0];
    const float* Wi    = (const float*)d_in[1];
    const float* bi    = (const float*)d_in[2];
    const float* enc   = (const float*)d_in[3];
    const float* enc_v = (const float*)d_in[4];
    const float* dec   = (const float*)d_in[5];
    const float* Wo    = (const float*)d_in[6];
    const float* bo    = (const float*)d_in[7];
    float* out = (float*)d_out;

    // ---- workspace carve -------------------------------------------------
    char* ws = (char*)d_ws;
    size_t off = 0;
    auto carve = [&](size_t bytes) -> char* {
        char* p = ws + off;
        off += (bytes + 255) & ~(size_t)255;
        return p;
    };
    __bf16* encb  = (__bf16*)carve((size_t)kNH * kD * kNI * 2);
    __bf16* encvb = (__bf16*)carve((size_t)kNH * kD * kNI * 2);
    __bf16* decb  = (__bf16*)carve((size_t)kNH * kNI * kD * 2);
    __bf16* hb    = (__bf16*)carve((size_t)kB * kT * kD * 2);
    float*  h32   = (float*) carve((size_t)kB * kT * kD * 4);
    float*  t32   = (float*) carve((size_t)kB * kT * kD * 4);
    __bf16* hsp   = (__bf16*)carve((size_t)kB * kNH * kT * kNI * 2);
    __bf16* qr    = (__bf16*)carve((size_t)kB * kNH * kT * kNI * 2);
    __bf16* sm    = (__bf16*)carve((size_t)kB * kNH * kT * kT * 2);
    float*  y32   = (float*) carve((size_t)kB * kNH * kT * kD * 4);
    __bf16* ykv   = (__bf16*)carve((size_t)kB * kNH * kT * kD * 2);
    __bf16* xy    = (__bf16*)carve((size_t)kB * kT * kNH * kNI * 2);

    // ---- weights -> bf16 -------------------------------------------------
    {
        const long ne = (long)kNH * kD * kNI;
        f2bf_kernel<<<dim3((unsigned)((ne + 255) / 256)), 256, 0, stream>>>(enc,   encb,  ne);
        f2bf_kernel<<<dim3((unsigned)((ne + 255) / 256)), 256, 0, stream>>>(enc_v, encvb, ne);
        f2bf_kernel<<<dim3((unsigned)((ne + 255) / 256)), 256, 0, stream>>>(dec,   decb,  ne);
    }

    // ---- input projection + LN ------------------------------------------
    inproj_kernel<<<kB * kT, 256, 0, stream>>>(x, Wi, bi, t32);
    ln512_kernel<<<kB * kT, 256, 0, stream>>>(t32, h32, hb);

    // ---- layers ----------------------------------------------------------
    for (int l = 0; l < kNL; ++l) {
        // (a) h_sp = relu(h @ enc[h])   per (b,h): [T,D] @ [D,N] -> bf16
        for (int b = 0; b < kB; ++b)
            for (int h = 0; h < kNH; ++h) {
                const __bf16* Ah = hb + (long)b * kT * kD;
                const __bf16* Bh = encb + (long)h * kD * kNI;
                __bf16* Ch = hsp + ((long)(b * kNH + h)) * kT * kNI;
                gemm128_wmma<1, false><<<dim3(kNI / BN, kT / BM), 256, SMEM_BYTES, stream>>>(
                    Ah, Bh, nullptr, Ch, nullptr,
                    kT, kNI, kD, kD, kNI, kNI, 0);
            }

        // (b) RoPE -> QR
        rope_kernel<<<dim3(kB * kNH * kT, (kNI / 2) / 256), 256, 0, stream>>>(hsp, qr);

        // (c) scores = (QR @ QR^T) * strict_tril    per (b,h)  [TDM both tiles]
        for (int bh = 0; bh < kB * kNH; ++bh) {
            const __bf16* Q = qr + (long)bh * kT * kNI;
            __bf16* S = sm + (long)bh * kT * kT;
            gemm128_wmma<3, true><<<dim3(kT / BN, kT / BM), 256, SMEM_BYTES, stream>>>(
                Q, Q, nullptr, S, nullptr,
                kT, kT, kNI, kNI, kNI, kT, 0);
        }

        // (d) Y = S @ V (V = h)   per (b,h): [T,T] @ [T,D] -> f32
        for (int b = 0; b < kB; ++b)
            for (int h = 0; h < kNH; ++h) {
                const int bh = b * kNH + h;
                const __bf16* S = sm + (long)bh * kT * kT;
                const __bf16* V = hb + (long)b * kT * kD;
                float* Y = y32 + (long)bh * kT * kD;
                gemm128_wmma<0, false><<<dim3(kD / BN, kT / BM), 256, SMEM_BYTES, stream>>>(
                    S, V, Y, nullptr, nullptr,
                    kT, kD, kT, kT, kD, kD, 0);
            }

        // (e) yKV = LN(Y) -> bf16 rows
        ln512_kernel<<<kB * kNH * kT, 256, 0, stream>>>(y32, nullptr, ykv);

        // (f) xy = h_sp * relu(yKV @ enc_v[h]) -> [B,T,NH*N] head-major flatten
        for (int b = 0; b < kB; ++b)
            for (int h = 0; h < kNH; ++h) {
                const int bh = b * kNH + h;
                const __bf16* Ah = ykv + (long)bh * kT * kD;
                const __bf16* Bh = encvb + (long)h * kD * kNI;
                const __bf16* G  = hsp + (long)bh * kT * kNI;
                __bf16* Ch = xy + (long)b * kT * (kNH * kNI) + (long)h * kNI;
                gemm128_wmma<2, false><<<dim3(kNI / BN, kT / BM), 256, SMEM_BYTES, stream>>>(
                    Ah, Bh, nullptr, Ch, G,
                    kT, kNI, kD, kD, kNI, kNH * kNI, kNI);
            }

        // (g) t = xy @ dec   per b: [T, NH*N] @ [NH*N, D] -> f32
        for (int b = 0; b < kB; ++b) {
            const __bf16* Ab = xy + (long)b * kT * (kNH * kNI);
            float* Cb = t32 + (long)b * kT * kD;
            gemm128_wmma<0, false><<<dim3(kD / BN, kT / BM), 256, SMEM_BYTES, stream>>>(
                Ab, decb, Cb, nullptr, nullptr,
                kT, kD, kNH * kNI, kNH * kNI, kD, kD, 0);
        }

        // (h) h = LN(h + LN(t))
        lnresln_kernel<<<kB * kT, 256, 0, stream>>>(t32, h32, hb);
    }

    // ---- output projection ----------------------------------------------
    outproj_kernel<<<kB * kT, 256, 0, stream>>>(h32, Wo, bo, out);
}
